// GCN_12652973654219
// MI455X (gfx1250) — matrix operand order repaired and checked by reading, not compiled, not run on hardware
//
#include <hip/hip_runtime.h>
#include <hip/hip_bf16.h>
#include <cfloat>

typedef __attribute__((ext_vector_type(16))) _Float16 v16h;
typedef __attribute__((ext_vector_type(8)))  _Float16 v8h_t;
typedef __attribute__((ext_vector_type(4)))  _Float16 v4h_t;
typedef __attribute__((ext_vector_type(8)))  float    v8f;

#define GCN_N 100000
#define GCN_E 1600000
#define GCN_G 512

static __device__ __forceinline__ float gelu_exact(float v) {
  return 0.5f * v * (1.0f + erff(v * 0.70710678118654752440f));
}

static __device__ __forceinline__ void atomic_max_f32(float* addr, float v) {
  if (v >= 0.0f) atomicMax((int*)addr, __float_as_int(v));
  else           atomicMin((unsigned int*)addr, __float_as_uint(v));
}

// ---------------- elementwise / setup kernels ----------------

__global__ void k_cvt_f16(const float* __restrict__ in, _Float16* __restrict__ out, int n) {
  int i = blockIdx.x * blockDim.x + threadIdx.x;
  if (i < n) out[i] = (_Float16)in[i];
}

__global__ void k_deg_init(float* __restrict__ deg, int n) {
  int i = blockIdx.x * blockDim.x + threadIdx.x;
  if (i < n) deg[i] = 1.0f;   // self-loop contribution
}

__global__ void k_deg_edges(const int* __restrict__ dst, float* __restrict__ deg, int e) {
  int i = blockIdx.x * blockDim.x + threadIdx.x;
  if (i < e) atomicAdd(&deg[dst[i]], 1.0f);
}

__global__ void k_dinv(const float* __restrict__ deg, float* __restrict__ dinv, int n) {
  int i = blockIdx.x * blockDim.x + threadIdx.x;
  if (i < n) { float d = deg[i]; dinv[i] = (d > 0.0f) ? rsqrtf(d) : 0.0f; }
}

__global__ void k_norm(const int* __restrict__ src, const int* __restrict__ dst,
                       const float* __restrict__ dinv, float* __restrict__ nrm, int e) {
  int i = blockIdx.x * blockDim.x + threadIdx.x;
  if (i < e) nrm[i] = dinv[src[i]] * dinv[dst[i]];
}

// ---------------- weight pack: W[K,DOUT] f32 -> B-fragment-ready f16 ----------------
// P[((tn*KT + kt)*32 + lane)*16 + e] = W[k, tn*16 + lane%16] with
//   k = kt*32 + (lane>=16 ? 16 : 0) + e   (zero-padded past K)
__global__ void k_pack_w(const float* __restrict__ W, _Float16* __restrict__ P, int K, int dout) {
  int KT = (K + 31) >> 5;
  int total = (dout >> 4) * KT * 512;
  int tid = blockIdx.x * blockDim.x + threadIdx.x;
  if (tid >= total) return;
  int e    = tid & 15;
  int lane = (tid >> 4) & 31;
  int kt   = (tid >> 9) % KT;
  int tn   = (tid >> 9) / KT;
  int col  = tn * 16 + (lane & 15);
  int k    = kt * 32 + ((lane >> 4) ? 16 : 0) + e;
  P[tid] = (k < K) ? (_Float16)W[(size_t)k * dout + col] : (_Float16)0.0f;
}

// ---------------- WMMA GEMM: C[N,DOUT] = A[N,K](f16) x Wpacked ----------------
// One wave per 16x16 tile; A via two b128 loads, B via two b128 loads from packed buffer.
template <int K, int DOUT>
__global__ void k_gemm_wmma(const _Float16* __restrict__ A, const _Float16* __restrict__ Bp,
                            float* __restrict__ C) {
  constexpr int KT = (K + 31) / 32;
  constexpr int tilesN = DOUT / 16;
  const int lane = threadIdx.x & 31;
  const int wave = blockIdx.x * (blockDim.x >> 5) + (threadIdx.x >> 5);
  const int numTiles = (GCN_N / 16) * tilesN;
  if (wave >= numTiles) return;                 // wave-uniform; EXEC stays all-1s
  const int tm  = wave / tilesN;
  const int tn  = wave - tm * tilesN;
  const int hi  = lane >> 4;
  const int row = tm * 16 + (lane & 15);
  const int col = tn * 16 + (lane & 15);

  const _Float16* arow = A + (size_t)row * K;
  const _Float16* bbase = Bp + ((size_t)tn * KT * 32 + lane) * 16;
  __builtin_prefetch(bbase, 0, 1);

  v8f acc = {};
#pragma unroll
  for (int kt = 0; kt < KT; ++kt) {
    v16h a;
    if constexpr (K == 2) {
      _Float16 a0 = arow[0];
      _Float16 a1 = arow[1];
#pragma unroll
      for (int j = 0; j < 16; ++j) a[j] = (_Float16)0.0f;
      a[0] = hi ? (_Float16)0.0f : a0;          // lanes 0-15 hold K=0,1
      a[1] = hi ? (_Float16)0.0f : a1;
    } else {
      const v8h_t* ap = (const v8h_t*)(arow + kt * 32 + hi * 16);
      v8h_t alo = ap[0];                         // K = kt*32 + hi*16 .. +7
      v8h_t ahi = ap[1];                         // K = kt*32 + hi*16 + 8 .. +15
#pragma unroll
      for (int j = 0; j < 8; ++j) { a[j] = alo[j]; a[8 + j] = ahi[j]; }
    }
    const v8h_t* bp = (const v8h_t*)(bbase + (size_t)kt * 512);
    v8h_t blo = bp[0];
    v8h_t bhi = bp[1];
    v16h b;
#pragma unroll
    for (int j = 0; j < 8; ++j) { b[j] = blo[j]; b[8 + j] = bhi[j]; }
    acc = __builtin_amdgcn_wmma_f32_16x16x32_f16(false, a, false, b, (short)0, acc, false, false);
  }
  const int mbase = tm * 16 + hi * 8;
#pragma unroll
  for (int v = 0; v < 8; ++v)
    C[(size_t)(mbase + v) * DOUT + col] = acc[v];
}

// ---------------- edge scatter-add (float4): agg[dst] += hw[src] * norm ----------------
__global__ void k_scatter4(const int* __restrict__ src, const int* __restrict__ dst,
                           const float* __restrict__ nrm, const float4* __restrict__ hw,
                           float* __restrict__ agg, int e, int ldq /* log2(dout/4) */) {
  int tid = blockIdx.x * blockDim.x + threadIdx.x;
  int total = e << ldq;
  if (tid >= total) return;
  int ei = tid >> ldq;
  int q  = tid & ((1 << ldq) - 1);
  int s = src[ei], d = dst[ei];
  float w = nrm[ei];
  float4 v = hw[((size_t)s << ldq) + q];
  float* base = agg + (((size_t)d << ldq) + q) * 4;
  atomicAdd(base + 0, v.x * w);
  atomicAdd(base + 1, v.y * w);
  atomicAdd(base + 2, v.z * w);
  atomicAdd(base + 3, v.w * w);
}

// ---------------- fused self-loop + bias + BN(eval) + GELU -> f16 (x4) ----------------
__global__ void k_post4(const float4* __restrict__ agg, const float4* __restrict__ hw,
                        const float* __restrict__ dinv, const float* __restrict__ bias,
                        const float* __restrict__ gamma, const float* __restrict__ beta,
                        v4h_t* __restrict__ act, int ldq) {
  int tid = blockIdx.x * blockDim.x + threadIdx.x;
  int total = GCN_N << ldq;
  if (tid >= total) return;
  int i  = tid >> ldq;
  int f0 = (tid & ((1 << ldq) - 1)) << 2;
  float di = dinv[i];
  float s = di * di;
  float4 va = agg[tid];
  float4 vh = hw[tid];
  float r[4] = {va.x + vh.x * s + bias[f0 + 0], va.y + vh.y * s + bias[f0 + 1],
                va.z + vh.z * s + bias[f0 + 2], va.w + vh.w * s + bias[f0 + 3]};
  if (gamma) {
    const float inv = 0.99999500003749978f;     // 1/sqrt(1+1e-5)
#pragma unroll
    for (int j = 0; j < 4; ++j) r[j] = r[j] * (gamma[f0 + j] * inv) + beta[f0 + j];
  }
  v4h_t o;
#pragma unroll
  for (int j = 0; j < 4; ++j) o[j] = (_Float16)gelu_exact(r[j]);
  act[tid] = o;
}

// ---------------- global max pool + head ----------------
__global__ void k_pool_init(float* __restrict__ pool, int n) {
  int i = blockIdx.x * blockDim.x + threadIdx.x;
  if (i < n) pool[i] = __int_as_float(0xff800000);  // -inf
}

__global__ void k_pool_max(const _Float16* __restrict__ act, const int* __restrict__ batch,
                           float* __restrict__ pool, int n /* N*64 */) {
  int tid = blockIdx.x * blockDim.x + threadIdx.x;
  if (tid >= n) return;
  int i = tid >> 6;
  int f = tid & 63;
  int g = batch[i];
  atomic_max_f32(&pool[(size_t)g * 64 + f], (float)act[tid]);
}

__global__ void k_head(const float* __restrict__ pool,
                       const float* __restrict__ lw1, const float* __restrict__ lb1,
                       const float* __restrict__ lw2, const float* __restrict__ lb2,
                       float* __restrict__ out) {
  int g = blockIdx.x * blockDim.x + threadIdx.x;
  if (g >= GCN_G) return;
  const float* p = pool + (size_t)g * 64;
  float t[10];
#pragma unroll
  for (int j = 0; j < 10; ++j) {
    float s = lb1[j];
    for (int i = 0; i < 64; ++i) s += p[i] * lw1[i * 10 + j];
    t[j] = s;
  }
#pragma unroll
  for (int c = 0; c < 2; ++c) {
    float s = lb2[c];
#pragma unroll
    for (int j = 0; j < 10; ++j) s += t[j] * lw2[j * 2 + c];
    out[(size_t)g * 2 + c] = s;
  }
}

// ---------------- host orchestration ----------------

static inline void* carve(char*& p, size_t bytes) {
  void* r = (void*)p;
  p += (bytes + 255) & ~(size_t)255;
  return r;
}

extern "C" void kernel_launch(void* const* d_in, const int* in_sizes, int n_in,
                              void* d_out, int out_size, void* d_ws, size_t ws_size,
                              hipStream_t stream) {
  (void)in_sizes; (void)n_in; (void)out_size; (void)ws_size;
  const float* x     = (const float*)d_in[0];
  const int*   ei    = (const int*)d_in[1];
  const int*   batch = (const int*)d_in[2];
  const float* W[5]  = {(const float*)d_in[3], (const float*)d_in[5], (const float*)d_in[7],
                        (const float*)d_in[9], (const float*)d_in[11]};
  const float* b[5]  = {(const float*)d_in[4], (const float*)d_in[6], (const float*)d_in[8],
                        (const float*)d_in[10], (const float*)d_in[12]};
  const float* g1 = (const float*)d_in[13]; const float* be1 = (const float*)d_in[14];
  const float* g2 = (const float*)d_in[15]; const float* be2 = (const float*)d_in[16];
  const float* g3 = (const float*)d_in[17]; const float* be3 = (const float*)d_in[18];
  const float* lw1 = (const float*)d_in[19]; const float* lb1 = (const float*)d_in[20];
  const float* lw2 = (const float*)d_in[21]; const float* lb2 = (const float*)d_in[22];
  float* out = (float*)d_out;

  const int* src = ei;            // ei[0]
  const int* dst = ei + GCN_E;    // ei[1]

  const int din_[5]  = {2, 64, 64, 128, 128};
  const int dout_[5] = {64, 64, 128, 128, 64};

  // workspace carve
  char* p = (char*)d_ws;
  float*    deg  = (float*)carve(p, (size_t)GCN_N * 4);
  float*    dinv = (float*)carve(p, (size_t)GCN_N * 4);
  float*    nrm  = (float*)carve(p, (size_t)GCN_E * 4);
  _Float16* wp[5];
  for (int l = 0; l < 5; ++l) {
    int KT = (din_[l] + 31) >> 5;
    wp[l] = (_Float16*)carve(p, (size_t)(dout_[l] / 16) * KT * 512 * 2);
  }
  _Float16* actA = (_Float16*)carve(p, (size_t)GCN_N * 128 * 2);
  _Float16* actB = (_Float16*)carve(p, (size_t)GCN_N * 128 * 2);
  float*    hw   = (float*)carve(p, (size_t)GCN_N * 128 * 4);
  float*    agg  = (float*)carve(p, (size_t)GCN_N * 128 * 4);
  float*    pool = (float*)carve(p, (size_t)GCN_G * 64 * 4);

  const int B256 = 256;
  auto blocks = [](long long n) { return (int)((n + 255) / 256); };

  // degrees / norm
  k_deg_init <<<blocks(GCN_N), B256, 0, stream>>>(deg, GCN_N);
  k_deg_edges<<<blocks(GCN_E), B256, 0, stream>>>(dst, deg, GCN_E);
  k_dinv     <<<blocks(GCN_N), B256, 0, stream>>>(deg, dinv, GCN_N);
  k_norm     <<<blocks(GCN_E), B256, 0, stream>>>(src, dst, dinv, nrm, GCN_E);

  // weight pack (f32 -> f16 fragment layout) + input conversion
  for (int l = 0; l < 5; ++l) {
    int KT = (din_[l] + 31) >> 5;
    int n = (dout_[l] / 16) * KT * 512;
    k_pack_w<<<blocks(n), B256, 0, stream>>>(W[l], wp[l], din_[l], dout_[l]);
  }
  k_cvt_f16<<<blocks(GCN_N * 2), B256, 0, stream>>>(x, actA, GCN_N * 2);

  const float* gam[5] = {g1, nullptr, g2, nullptr, g3};
  const float* bet[5] = {be1, nullptr, be2, nullptr, be3};
  _Float16* aIn = actA;
  _Float16* aOut = actB;

  for (int l = 0; l < 5; ++l) {
    const int dO = dout_[l];
    const int ldq = (dO == 64) ? 4 : 5;          // log2(dout/4)
    const int tiles = (GCN_N / 16) * (dO / 16);
    const int gemmBlocks = (tiles + 7) / 8;
    switch (l) {
      case 0: k_gemm_wmma<2,   64 ><<<gemmBlocks, 256, 0, stream>>>(aIn, wp[l], hw); break;
      case 1: k_gemm_wmma<64,  64 ><<<gemmBlocks, 256, 0, stream>>>(aIn, wp[l], hw); break;
      case 2: k_gemm_wmma<64,  128><<<gemmBlocks, 256, 0, stream>>>(aIn, wp[l], hw); break;
      case 3: k_gemm_wmma<128, 128><<<gemmBlocks, 256, 0, stream>>>(aIn, wp[l], hw); break;
      case 4: k_gemm_wmma<128, 64 ><<<gemmBlocks, 256, 0, stream>>>(aIn, wp[l], hw); break;
    }
    hipMemsetAsync(agg, 0, (size_t)GCN_N * dO * sizeof(float), stream);
    long long totalE = (long long)GCN_E << ldq;
    k_scatter4<<<blocks(totalE), B256, 0, stream>>>(src, dst, nrm, (const float4*)hw, agg,
                                                    GCN_E, ldq);
    long long totalN = (long long)GCN_N << ldq;
    k_post4<<<blocks(totalN), B256, 0, stream>>>((const float4*)agg, (const float4*)hw, dinv,
                                                 b[l], gam[l], bet[l], (v4h_t*)aOut, ldq);
    _Float16* t = aIn; aIn = aOut; aOut = t;
  }
  // after 5 swaps the final [N,64] f16 activation is in aIn

  k_pool_init<<<blocks(GCN_G * 64), B256, 0, stream>>>(pool, GCN_G * 64);
  k_pool_max <<<blocks((long long)GCN_N * 64), B256, 0, stream>>>(aIn, batch, pool, GCN_N * 64);
  k_head     <<<(GCN_G + 255) / 256, B256, 0, stream>>>(pool, lw1, lb1, lw2, lb2, out);
}
